// GATCoder_29180007809580
// MI455X (gfx1250) — compile-verified, matrix operand order checked by reading, not run on hardware
//
#include <hip/hip_runtime.h>
#include <hip/hip_bf16.h>
#include <math.h>

// Problem constants (from reference)
#define NNODES 50000
#define NEDGES 800000
#define IN_F   256
#define HID    32
#define HEADS  4
#define OUT_F  128
#define F1     (HEADS * HID)   // 128
#define NEG_SLOPE 0.2f

typedef __attribute__((ext_vector_type(16))) __bf16 bf16x16;
typedef __attribute__((ext_vector_type(8)))  float  f32x8;

__device__ __forceinline__ unsigned short f32_to_bf16_rne(float f) {
  unsigned u = __float_as_uint(f);
  unsigned r = u + 0x7FFFu + ((u >> 16) & 1u);   // round-to-nearest-even
  return (unsigned short)(r >> 16);
}

// ---------------------------------------------------------------------------
// Pack W [K x 128] f32 row-major into WMMA B-fragments (bf16), per-lane
// contiguous: frag f = kt*8+nt ; bf16 element ((f*32+lane)*16 + i).
// K mapping per 16-bit fragment layout: lane half h: i<8 -> k=i+8h ; i>=8 -> k=i+8+8h.
// ---------------------------------------------------------------------------
__global__ void pack_w_frags(const float* __restrict__ W, unsigned short* __restrict__ P,
                             int K, int OUTC) {
  int t = blockIdx.x * blockDim.x + threadIdx.x;
  int numNT = OUTC >> 4;
  int nFrag = (K >> 5) * numNT;
  if (t >= nFrag * 32) return;
  int lane = t & 31;
  int frag = t >> 5;
  int nt = frag % numNT;
  int kt = frag / numNT;
  int half = lane >> 4;
  int n = nt * 16 + (lane & 15);
  unsigned short* dstp = P + (size_t)t * 16;
#pragma unroll
  for (int i = 0; i < 16; ++i) {
    int k = kt * 32 + ((i < 8) ? (i + half * 8) : (i + 8 + half * 8));
    dstp[i] = f32_to_bf16_rne(W[(size_t)k * OUTC + n]);
  }
}

// Raw (pre-conversion) A fragment: 16 f32 in 4 vectors
struct ARaw { float4 r0, r1, r2, r3; };

__device__ __forceinline__ ARaw load_a_raw(const float* __restrict__ ap) {
  ARaw f;
  f.r0 = *(const float4*)(ap + 0);    // k +0..3
  f.r1 = *(const float4*)(ap + 4);    // k +4..7
  f.r2 = *(const float4*)(ap + 16);   // k +16..19
  f.r3 = *(const float4*)(ap + 20);   // k +20..23
  return f;
}

// Hardware v_cvt_pk_bf16_f32 path
__device__ __forceinline__ bf16x16 cvt_a(const ARaw& f) {
  bf16x16 a;
  a[0]  = (__bf16)f.r0.x; a[1]  = (__bf16)f.r0.y; a[2]  = (__bf16)f.r0.z; a[3]  = (__bf16)f.r0.w;
  a[4]  = (__bf16)f.r1.x; a[5]  = (__bf16)f.r1.y; a[6]  = (__bf16)f.r1.z; a[7]  = (__bf16)f.r1.w;
  a[8]  = (__bf16)f.r2.x; a[9]  = (__bf16)f.r2.y; a[10] = (__bf16)f.r2.z; a[11] = (__bf16)f.r2.w;
  a[12] = (__bf16)f.r3.x; a[13] = (__bf16)f.r3.y; a[14] = (__bf16)f.r3.z; a[15] = (__bf16)f.r3.w;
  return a;
}

// ---------------------------------------------------------------------------
// Y[M,128] = X[M,K] @ W via v_wmma_f32_16x16x32_bf16.
// Packed W staged in LDS once per block (cooperative copy, all 4 waves).
// One wave owns 16 rows x 128 cols: 8 f32 accumulators, 8 WMMA per k-step.
// Structural software pipeline: issue k+1 A (raw f32) + B (LDS) loads before
// the 8 WMMAs of step k, so load waits hide behind a full matrix block.
// ---------------------------------------------------------------------------
template <int K>
__global__ __launch_bounds__(128, 1)
void gemm_wmma_bf16(const float* __restrict__ X, const unsigned short* __restrict__ P,
                    float* __restrict__ Y, int M) {
  constexpr int OUTC = 128;
  constexpr int KT = K / 32;
  constexpr int LDS_BYTES = KT * 8 * 32 * 32;   // frags * lanes * 32B
  extern __shared__ char smem[];

  // cooperative copy of packed W into LDS (b128 chunks)
  {
    const float4* gsrc = (const float4*)P;
    float4* ldst = (float4*)smem;
    const int nChunks = LDS_BYTES / 16;
    for (int i = threadIdx.x; i < nChunks; i += 128) ldst[i] = gsrc[i];
  }
  __syncthreads();

  const int lane = threadIdx.x & 31;
  const int wave = threadIdx.x >> 5;
  const int half = lane >> 4;
  const int l15  = lane & 15;
  const int rowTile = (blockIdx.x * 4 + wave) << 4;
  if (rowTile >= M) return;          // uniform per wave (after the barrier)
  const int row = rowTile + l15;

  const bf16x16* Bf = (const bf16x16*)smem;
  f32x8 c[8] = {};
  const float* xrow = X + (size_t)row * K + half * 8;

  // prologue: stage k=0 fragments
  ARaw araw = load_a_raw(xrow);
  bf16x16 bcur[8];
#pragma unroll
  for (int nt = 0; nt < 8; ++nt) bcur[nt] = Bf[(size_t)nt * 32 + lane];

#pragma unroll
  for (int kt = 0; kt < KT; ++kt) {
    const int ktn = (kt + 1 < KT) ? (kt + 1) : kt;   // clamp (harmless reload at tail)
    // issue next-step loads first: 4x global_load_b128 + 16x ds_load_b128
    ARaw anext = load_a_raw(xrow + ktn * 32);
    bf16x16 bnext[8];
#pragma unroll
    for (int nt = 0; nt < 8; ++nt)
      bnext[nt] = Bf[(size_t)(ktn * 8 + nt) * 32 + lane];

    // convert current A, then 8 back-to-back WMMAs on the XDL pipe
    bf16x16 a = cvt_a(araw);
#pragma unroll
    for (int nt = 0; nt < 8; ++nt)
      c[nt] = __builtin_amdgcn_wmma_f32_16x16x32_bf16(false, a, false, bcur[nt],
                                                      (short)0, c[nt], false, false);

    araw = anext;
#pragma unroll
    for (int nt = 0; nt < 8; ++nt) bcur[nt] = bnext[nt];
  }

  // C/D layout: lane(0-15)=col n within tile, VGPR r = row r; lanes 16-31 = rows 8..15
  float* yb = Y + (size_t)(rowTile + half * 8) * OUTC + l15;
#pragma unroll
  for (int nt = 0; nt < 8; ++nt) {
#pragma unroll
    for (int r = 0; r < 8; ++r) {
      yb[(size_t)r * OUTC + nt * 16] = c[nt][r];
    }
  }
}

// el/er per (node, head): dot(ft[n,h,:], attn[h,:])
__global__ void attn_dots(const float* __restrict__ ft, const float* __restrict__ al,
                          const float* __restrict__ ar, float* __restrict__ el,
                          float* __restrict__ er, int H, int D) {
  int t = blockIdx.x * blockDim.x + threadIdx.x;
  if (t >= NNODES * H) return;
  int h = t % H;
  const float* f = ft + (size_t)t * D;   // ft is [N, H, D] contiguous
  const float* alh = al + h * D;
  const float* arh = ar + h * D;
  float sl = 0.f, sr = 0.f;
  for (int d = 0; d < D; d += 4) {
    float4 v = *(const float4*)(f + d);
    float4 l = *(const float4*)(alh + d);
    float4 r = *(const float4*)(arh + d);
    sl += v.x * l.x + v.y * l.y + v.z * l.z + v.w * l.w;
    sr += v.x * r.x + v.y * r.y + v.z * r.z + v.w * r.w;
  }
  el[t] = sl;
  er[t] = sr;
}

__global__ void init_accum(float* __restrict__ emax, float* __restrict__ denom,
                           float* __restrict__ acc, int H, int D) {
  int t = blockIdx.x * blockDim.x + threadIdx.x;
  int nh = NNODES * H;
  if (t < nh) { emax[t] = -INFINITY; denom[t] = 0.f; }
  if (t < nh * D) acc[t] = 0.f;
}

// float atomic max via sign-split integer atomics (init must be -inf)
__device__ __forceinline__ void atomicMaxF(float* addr, float val) {
  if (!(val < 0.f)) atomicMax((int*)addr, __float_as_int(val));
  else              atomicMin((unsigned int*)addr, __float_as_uint(val));
}

__device__ __forceinline__ float lrelu(float x) {
  return x > 0.f ? x : NEG_SLOPE * x;
}

__global__ void edge_pass_max(const int* __restrict__ src, const int* __restrict__ dst,
                              const float* __restrict__ el, const float* __restrict__ er,
                              float* __restrict__ emax, int H) {
  int t = blockIdx.x * blockDim.x + threadIdx.x;
  if (t >= NEDGES * H) return;
  int e = t / H, h = t - e * H;
  int s = src[e], d = dst[e];
  float x = lrelu(el[s * H + h] + er[d * H + h]);
  atomicMaxF(&emax[d * H + h], x);
}

__global__ void edge_pass_sum(const int* __restrict__ src, const int* __restrict__ dst,
                              const float* __restrict__ el, const float* __restrict__ er,
                              const float* __restrict__ emax, float* __restrict__ denom,
                              int H) {
  int t = blockIdx.x * blockDim.x + threadIdx.x;
  if (t >= NEDGES * H) return;
  int e = t / H, h = t - e * H;
  int s = src[e], d = dst[e];
  float x = lrelu(el[s * H + h] + er[d * H + h]);
  float m = emax[d * H + h];
  if (m == -INFINITY || m != m) m = 0.f;
  atomicAdd(&denom[d * H + h], __expf(x - m));
}

__global__ void edge_pass_scatter(const int* __restrict__ src, const int* __restrict__ dst,
                                  const float* __restrict__ el, const float* __restrict__ er,
                                  const float* __restrict__ emax, const float* __restrict__ denom,
                                  const float* __restrict__ ft, float* __restrict__ acc,
                                  int H, int D) {
  int t = blockIdx.x * blockDim.x + threadIdx.x;
  if (t >= NEDGES * H) return;
  int e = t / H, h = t - e * H;
  int s = src[e], d = dst[e];
  float x = lrelu(el[s * H + h] + er[d * H + h]);
  float m = emax[d * H + h];
  if (m == -INFINITY || m != m) m = 0.f;
  float dn = denom[d * H + h];
  if (dn == 0.f) dn = 1.f;
  float alpha = __expf(x - m) / dn;
  const float* fs = ft  + ((size_t)s * H + h) * D;
  float*       ad = acc + ((size_t)d * H + h) * D;
  __builtin_prefetch(fs, 0, 0);   // global_prefetch_b8 on the gathered row
  for (int k = 0; k < D; k += 4) {
    float4 v = *(const float4*)(fs + k);   // b128 gather (L2-resident row)
    atomicAdd(&ad[k + 0], alpha * v.x);
    atomicAdd(&ad[k + 1], alpha * v.y);
    atomicAdd(&ad[k + 2], alpha * v.z);
    atomicAdd(&ad[k + 3], alpha * v.w);
  }
}

__global__ void finalize(const float* __restrict__ acc, const float* __restrict__ bias,
                         float* __restrict__ out, int F, int do_elu) {
  int t = blockIdx.x * blockDim.x + threadIdx.x;
  if (t >= NNODES * F) return;
  int c = t % F;
  float v = acc[t] + bias[c];
  if (do_elu) v = v > 0.f ? v : __expf(v) - 1.f;
  out[t] = v;
}

// ---------------------------------------------------------------------------
extern "C" void kernel_launch(void* const* d_in, const int* in_sizes, int n_in,
                              void* d_out, int out_size, void* d_ws, size_t ws_size,
                              hipStream_t stream) {
  (void)in_sizes; (void)n_in; (void)out_size; (void)ws_size;
  const float* feat = (const float*)d_in[0];
  const int*   src  = (const int*)d_in[1];
  const int*   dst  = (const int*)d_in[2];
  const float* W1   = (const float*)d_in[3];
  const float* al1  = (const float*)d_in[4];
  const float* ar1  = (const float*)d_in[5];
  const float* b1   = (const float*)d_in[6];
  const float* W2   = (const float*)d_in[7];
  const float* al2  = (const float*)d_in[8];
  const float* ar2  = (const float*)d_in[9];
  const float* b2   = (const float*)d_in[10];
  float* out = (float*)d_out;

  // workspace carve-out (256B aligned slices)
  char* base = (char*)d_ws;
  size_t off = 0;
  auto alloc = [&](size_t bytes) -> void* {
    void* p = base + off;
    off += (bytes + 255) & ~(size_t)255;
    return p;
  };
  unsigned short* P1 = (unsigned short*)alloc((size_t)(IN_F / 32) * (F1 / 16) * 32 * 16 * 2);
  unsigned short* P2 = (unsigned short*)alloc((size_t)(F1 / 32) * (OUT_F / 16) * 32 * 16 * 2);
  float* ft    = (float*)alloc((size_t)NNODES * F1 * sizeof(float));
  float* acc   = (float*)alloc((size_t)NNODES * F1 * sizeof(float));
  float* hbuf  = (float*)alloc((size_t)NNODES * F1 * sizeof(float));
  float* el    = (float*)alloc((size_t)NNODES * HEADS * sizeof(float));
  float* er    = (float*)alloc((size_t)NNODES * HEADS * sizeof(float));
  float* emax  = (float*)alloc((size_t)NNODES * HEADS * sizeof(float));
  float* denom = (float*)alloc((size_t)NNODES * HEADS * sizeof(float));

  const int TB = 256;
  const int rowTiles = NNODES / 16;                  // 3125
  const int gemmBlocks = (rowTiles + 3) / 4;         // 4 waves (64 rows) per block
  const size_t LDS1 = (size_t)(IN_F / 32) * 8 * 32 * 32;  // 64 KB
  const size_t LDS2 = (size_t)(F1   / 32) * 8 * 32 * 32;  // 32 KB

  // pack weights into WMMA B-fragments (bf16)
  { int nT = (IN_F / 32) * (F1 / 16) * 32;
    pack_w_frags<<<(nT + TB - 1) / TB, TB, 0, stream>>>(W1, P1, IN_F, F1); }
  { int nT = (F1 / 32) * (OUT_F / 16) * 32;
    pack_w_frags<<<(nT + TB - 1) / TB, TB, 0, stream>>>(W2, P2, F1, OUT_F); }

  // ---------------- Layer 1 (H=4, D=32) ----------------
  gemm_wmma_bf16<IN_F><<<gemmBlocks, 128, LDS1, stream>>>(feat, P1, ft, NNODES);
  { int n = NNODES * HEADS;
    attn_dots<<<(n + TB - 1) / TB, TB, 0, stream>>>(ft, al1, ar1, el, er, HEADS, HID); }
  { int n = NNODES * HEADS * HID;
    init_accum<<<(n + TB - 1) / TB, TB, 0, stream>>>(emax, denom, acc, HEADS, HID); }
  { int n = NEDGES * HEADS;
    edge_pass_max    <<<(n + TB - 1) / TB, TB, 0, stream>>>(src, dst, el, er, emax, HEADS);
    edge_pass_sum    <<<(n + TB - 1) / TB, TB, 0, stream>>>(src, dst, el, er, emax, denom, HEADS);
    edge_pass_scatter<<<(n + TB - 1) / TB, TB, 0, stream>>>(src, dst, el, er, emax, denom,
                                                            ft, acc, HEADS, HID); }
  { int n = NNODES * F1;
    finalize<<<(n + TB - 1) / TB, TB, 0, stream>>>(acc, b1, hbuf, F1, 1); }

  // ---------------- Layer 2 (H=1, D=128) ----------------
  gemm_wmma_bf16<F1><<<gemmBlocks, 128, LDS2, stream>>>(hbuf, P2, ft, NNODES);
  { int n = NNODES;
    attn_dots<<<(n + TB - 1) / TB, TB, 0, stream>>>(ft, al2, ar2, el, er, 1, OUT_F); }
  { int n = NNODES * OUT_F;
    init_accum<<<(n + TB - 1) / TB, TB, 0, stream>>>(emax, denom, acc, 1, OUT_F); }
  { int n = NEDGES;
    edge_pass_max    <<<(n + TB - 1) / TB, TB, 0, stream>>>(src, dst, el, er, emax, 1);
    edge_pass_sum    <<<(n + TB - 1) / TB, TB, 0, stream>>>(src, dst, el, er, emax, denom, 1);
    edge_pass_scatter<<<(n + TB - 1) / TB, TB, 0, stream>>>(src, dst, el, er, emax, denom,
                                                            ft, acc, 1, OUT_F); }
  { int n = NNODES * OUT_F;
    finalize<<<(n + TB - 1) / TB, TB, 0, stream>>>(acc, b2, out, OUT_F, 0); }
}
